// TransLayer_23579370455069
// MI455X (gfx1250) — matrix operand order repaired and checked by reading, not compiled, not run on hardware
//
#include <hip/hip_runtime.h>
#include <hip/hip_bf16.h>

// ===========================================================================
// Nystromformer layer for MI455X (gfx1250, wave32, WMMA).
// All matmuls -> v_wmma_f32_16x16x32_f16 (f16 in, f32 accumulate), all GEMMs
// in NT form (contiguous 32B B-fragments); transposed operands are produced
// for free in the preceding kernel's epilogue. out = a1 @ (pinv(a2) @ (a3@v))
// re-association removes [n x m] materialization; both softmax products use a
// fused flash-style kernel (online softmax, WMMA scores + WMMA PV).
// ===========================================================================

typedef _Float16 h16;
typedef __attribute__((ext_vector_type(16))) _Float16 v16h;
typedef __attribute__((ext_vector_type(8)))  _Float16 v8h;
typedef __attribute__((ext_vector_type(8)))  float    v8f;

#define DEVI static __device__ __forceinline__

static const int N_    = 4096;
static const int M_    = 256;   // landmarks
static const int BH_   = 32;    // b*h
static const int BN_   = 16384; // b*n

DEVI v8f wmma32(v16h a, v16h b, v8f c) {
  // D = A(16x32 f16) * B(32x16 f16) + C(16x16 f32)
  return __builtin_amdgcn_wmma_f32_16x16x32_f16(false, a, false, b, (short)0, c,
                                                false, false);
}

// A fragment, row-major A[M x K] contiguous in K (contiguous k order, matching load_b_nt):
// lane: m = lane&15, hi = lane>>4; halves 0..15 -> k0+16*hi.. (contiguous)
DEVI v16h load_a_nt(const h16* A, int lda, int m0, int k0) {
  int lane = threadIdx.x & 31;
  int hi = lane >> 4;
  const h16* p = A + (size_t)(m0 + (lane & 15)) * lda + k0;
  v8h lo = *(const v8h*)(p + 16 * hi);
  v8h hh = *(const v8h*)(p + 16 * hi + 8);
  v16h r;
#pragma unroll
  for (int i = 0; i < 8; ++i) { r[i] = lo[i]; r[i + 8] = hh[i]; }
  return r;
}

// B fragment for C = A @ Bm^T with Bm[N x K] row-major (NT gemm):
// lane: n = lane&15, hi = lane>>4; 16 contiguous halves at k0+16*hi.
DEVI v16h load_b_nt(const h16* Bm, int ldb, int n0, int k0) {
  int lane = threadIdx.x & 31;
  int hi = lane >> 4;
  const h16* p = Bm + (size_t)(n0 + (lane & 15)) * ldb + k0 + 16 * hi;
  return *(const v16h*)p;
}

// ---------------------------------------------------------------------------
// LayerNorm(x) -> f16                                        grid BN_, 256thr
// ---------------------------------------------------------------------------
__global__ void k_ln(const float* __restrict__ x, const float* __restrict__ w,
                     const float* __restrict__ bb, h16* __restrict__ nx) {
  int row = blockIdx.x, t = threadIdx.x;
  __shared__ float s1[256], s2[256];
  float a0 = x[row * 512 + t], a1 = x[row * 512 + t + 256];
  s1[t] = a0 + a1;
  s2[t] = a0 * a0 + a1 * a1;
  __syncthreads();
  for (int s = 128; s > 0; s >>= 1) {
    if (t < s) { s1[t] += s1[t + s]; s2[t] += s2[t + s]; }
    __syncthreads();
  }
  float mu = s1[0] * (1.0f / 512.0f);
  float var = s2[0] * (1.0f / 512.0f) - mu * mu;
  float ri = rsqrtf(var + 1e-5f);
  nx[row * 512 + t]       = (h16)((a0 - mu) * ri * w[t] + bb[t]);
  nx[row * 512 + t + 256] = (h16)((a1 - mu) * ri * w[t + 256] + bb[t + 256]);
}

// transpose+cast weight: out[c*rows + r] = (h16)in[r*cols + c]
__global__ void k_castT(const float* __restrict__ in, h16* __restrict__ out,
                        int rows, int cols) {
  int idx = blockIdx.x * 256 + threadIdx.x;
  int r = idx / cols, c = idx % cols;
  out[(size_t)c * rows + r] = (h16)in[idx];
}

// ---------------------------------------------------------------------------
// QKV GEMM: [16384 x 512] @ [512 x 1536] (NT, weights pre-transposed).
// One wave computes a 16x64 strip (4 accumulators share each A fragment).
// Epilogue scatters to q(scaled)/k (f16, [bh][n][64]), v (f32 + vT f16).
// grid (1024, 24), 32 threads
// ---------------------------------------------------------------------------
__global__ void k_gemm_qkv(const h16* __restrict__ A, const h16* __restrict__ BT,
                           h16* __restrict__ qf, h16* __restrict__ kf,
                           float* __restrict__ v32, h16* __restrict__ vT) {
  int m0 = blockIdx.x * 16, n0 = blockIdx.y * 64;
  v8f acc[4] = {v8f{}, v8f{}, v8f{}, v8f{}};
#pragma unroll 2
  for (int k0 = 0; k0 < 512; k0 += 32) {
    __builtin_prefetch(A + (size_t)m0 * 512 + k0 + 128, 0, 0);
    v16h af = load_a_nt(A, 512, m0, k0);
#pragma unroll
    for (int t = 0; t < 4; ++t)
      acc[t] = wmma32(af, load_b_nt(BT, 512, n0 + 16 * t, k0), acc[t]);
  }
  int lane = threadIdx.x & 31, hi = lane >> 4;
#pragma unroll
  for (int t = 0; t < 4; ++t) {
    int n = n0 + 16 * t + (lane & 15);
    int which = n / 512, rem = n % 512, hh = rem / 64, dd = rem % 64;
#pragma unroll
    for (int r = 0; r < 8; ++r) {
      int tok = m0 + r + 8 * hi;
      int bidx = tok >> 12, ii = tok & 4095;
      int bh = bidx * 8 + hh;
      size_t idx = ((size_t)bh * 4096 + ii) * 64 + dd;
      float val = acc[t][r];
      if (which == 0) {
        qf[idx] = (h16)(val * 0.125f);          // SCALE = 64^-0.5
      } else if (which == 1) {
        kf[idx] = (h16)val;
      } else {
        v32[idx] = val;
        vT[((size_t)bh * 64 + dd) * 4096 + ii] = (h16)val;
      }
    }
  }
}

// landmark means: [bh][256][64] from [bh][4096][64], group of 16 tokens
__global__ void k_landmarks(const h16* __restrict__ qf, const h16* __restrict__ kf,
                            h16* __restrict__ ql, h16* __restrict__ kl) {
  int idx = blockIdx.x * 256 + threadIdx.x;       // 32*256*64
  int dd = idx & 63, mm = (idx >> 6) & 255, bh = idx >> 14;
  float sq = 0.f, sk = 0.f;
  const h16* qp = qf + ((size_t)bh * 4096 + mm * 16) * 64 + dd;
  const h16* kp = kf + ((size_t)bh * 4096 + mm * 16) * 64 + dd;
#pragma unroll
  for (int j = 0; j < 16; ++j) { sq += (float)qp[j * 64]; sk += (float)kp[j * 64]; }
  ql[idx] = (h16)(sq * (1.f / 16.f));
  kl[idx] = (h16)(sk * (1.f / 16.f));
}

// sim2 = ql @ kl^T (NT, K=64), f32 out.  grid (16,16,32), 32 thr
__global__ void k_sim2(const h16* __restrict__ ql, const h16* __restrict__ kl,
                       float* __restrict__ sim2) {
  int bh = blockIdx.z;
  const h16* a = ql + (size_t)bh * 256 * 64;
  const h16* b = kl + (size_t)bh * 256 * 64;
  int m0 = blockIdx.x * 16, n0 = blockIdx.y * 16;
  v8f acc = {};
  acc = wmma32(load_a_nt(a, 64, m0, 0),  load_b_nt(b, 64, n0, 0),  acc);
  acc = wmma32(load_a_nt(a, 64, m0, 32), load_b_nt(b, 64, n0, 32), acc);
  int lane = threadIdx.x & 31, hi = lane >> 4, n = n0 + (lane & 15);
  float* c = sim2 + (size_t)bh * 256 * 256;
#pragma unroll
  for (int r = 0; r < 8; ++r) c[(size_t)(m0 + r + 8 * hi) * 256 + n] = acc[r];
}

// row softmax of sim2 -> a2 f16.  grid 8192 (=32*256 rows), 256 thr
__global__ void k_softmax2(const float* __restrict__ sim2, h16* __restrict__ a2) {
  int row = blockIdx.x, t = threadIdx.x;
  __shared__ float red[256];
  float v = sim2[(size_t)row * 256 + t];
  red[t] = v; __syncthreads();
  for (int s = 128; s > 0; s >>= 1) {
    if (t < s) red[t] = fmaxf(red[t], red[t + s]);
    __syncthreads();
  }
  float mx = red[0]; __syncthreads();
  float e = __expf(v - mx);
  red[t] = e; __syncthreads();
  for (int s = 128; s > 0; s >>= 1) {
    if (t < s) red[t] += red[t + s];
    __syncthreads();
  }
  a2[(size_t)row * 256 + t] = (h16)(e / red[0]);
}

// denom[bh] = max(rowsum)*max(colsum) of |a2| (a2>0).  grid 32, 256 thr
__global__ void k_denom(const h16* __restrict__ a2, float* __restrict__ denom) {
  int bh = blockIdx.x, t = threadIdx.x;
  const h16* m = a2 + (size_t)bh * 256 * 256;
  float rs = 0.f, cs = 0.f;
  for (int j = 0; j < 256; ++j) {
    rs += (float)m[(size_t)t * 256 + j];
    cs += (float)m[(size_t)j * 256 + t];
  }
  __shared__ float r1[256], r2[256];
  r1[t] = rs; r2[t] = cs; __syncthreads();
  for (int s = 128; s > 0; s >>= 1) {
    if (t < s) { r1[t] = fmaxf(r1[t], r1[t + s]); r2[t] = fmaxf(r2[t], r2[t + s]); }
    __syncthreads();
  }
  if (t == 0) denom[bh] = r1[0] * r2[0];
}

// z0 = a2^T / denom; also writes z0^T.  grid 8192, 256 thr
__global__ void k_z0(const h16* __restrict__ a2, const float* __restrict__ denom,
                     h16* __restrict__ z, h16* __restrict__ zT) {
  int idx = blockIdx.x * 256 + threadIdx.x;     // 32*256*256
  int bh = idx >> 16, j = idx & 255, i = (idx >> 8) & 255;
  float val = (float)a2[((size_t)bh * 256 + j) * 256 + i] / denom[bh];
  z[idx] = (h16)val;
  zT[((size_t)bh * 256 + j) * 256 + i] = (h16)val;
}

// ---------------------------------------------------------------------------
// Batched 256x256x256 NT WMMA GEMM for Newton-Schulz pinv.
// val = s1*acc + d1*I:  O1 (row-major, opt), O1T (transposed, opt)
// second output O2T = (s2*acc + d2*I)^T (opt) -- identity is symmetric.
// grid (16,16,32), 32 thr
// ---------------------------------------------------------------------------
__global__ void k_pinv_gemm(const h16* __restrict__ A, const h16* __restrict__ BT,
                            h16* __restrict__ O1, h16* __restrict__ O1T,
                            float s1, float d1,
                            h16* __restrict__ O2T, float s2, float d2) {
  int bh = blockIdx.z;
  const h16* a = A + (size_t)bh * 65536;
  const h16* b = BT + (size_t)bh * 65536;
  int m0 = blockIdx.x * 16, n0 = blockIdx.y * 16;
  v8f acc = {};
#pragma unroll 2
  for (int k0 = 0; k0 < 256; k0 += 32) {
    v16h af = load_a_nt(a, 256, m0, k0);
    v16h bf = load_b_nt(b, 256, n0, k0);
    acc = wmma32(af, bf, acc);
  }
  int lane = threadIdx.x & 31, hi = lane >> 4, n = n0 + (lane & 15);
#pragma unroll
  for (int r = 0; r < 8; ++r) {
    int m = m0 + r + 8 * hi;
    float id = (m == n) ? 1.f : 0.f;
    size_t base = (size_t)bh * 65536;
    h16 v1 = (h16)(s1 * acc[r] + d1 * id);
    if (O1)  O1[base + (size_t)m * 256 + n] = v1;
    if (O1T) O1T[base + (size_t)n * 256 + m] = v1;
    if (O2T) O2T[base + (size_t)n * 256 + m] = (h16)(s2 * acc[r] + d2 * id);
  }
}

// W^T = (z @ a3v)^T  (NT: B = a3v^T from flash).  grid (16,4,32), 32 thr
__global__ void k_w_gemm(const h16* __restrict__ Z, const h16* __restrict__ AVT,
                         h16* __restrict__ WT) {
  int bh = blockIdx.z;
  const h16* a = Z + (size_t)bh * 65536;
  const h16* b = AVT + (size_t)bh * 16384;    // [64][256]
  int m0 = blockIdx.x * 16, n0 = blockIdx.y * 16;
  v8f acc = {};
#pragma unroll 2
  for (int k0 = 0; k0 < 256; k0 += 32) {
    v16h af = load_a_nt(a, 256, m0, k0);
    v16h bf = load_b_nt(b, 256, n0, k0);
    acc = wmma32(af, bf, acc);
  }
  int lane = threadIdx.x & 31, hi = lane >> 4, n = n0 + (lane & 15);
#pragma unroll
  for (int r = 0; r < 8; ++r) {
    int m = m0 + r + 8 * hi;
    WT[(size_t)bh * 16384 + (size_t)n * 256 + m] = (h16)acc[r];
  }
}

// ---------------------------------------------------------------------------
// Fused  softmax(Q @ K^T) @ V  with online softmax. One wave per 16 Q-rows.
// Q [bh][nq][64] f16 (pre-scaled), K [bh][nk][64] f16, VT [bh][64][nk] f16.
// O f32 [bh][nq][64] (opt); OT f16 [bh][64][nq] (opt, transposed for NT use).
// grid (nq/16, BH), 32 thr
// ---------------------------------------------------------------------------
__global__ void k_flash(const h16* __restrict__ Q, const h16* __restrict__ K,
                        const h16* __restrict__ VT, float* __restrict__ O,
                        h16* __restrict__ OT, int nq, int nk) {
  int bh = blockIdx.y;
  const h16* q  = Q  + (size_t)bh * nq * 64;
  const h16* k  = K  + (size_t)bh * nk * 64;
  const h16* vt = VT + (size_t)bh * 64 * nk;
  int m0 = blockIdx.x * 16;
  int lane = threadIdx.x & 31, hi = lane >> 4, ln = lane & 15;
  __shared__ h16 lds_p[16][32];

  v16h aq0 = load_a_nt(q, 64, m0, 0);
  v16h aq1 = load_a_nt(q, 64, m0, 32);
  v8f acc[4] = {v8f{}, v8f{}, v8f{}, v8f{}};
  float mrow[8], lrow[8];
#pragma unroll
  for (int r = 0; r < 8; ++r) { mrow[r] = -1e30f; lrow[r] = 0.f; }

  for (int c0 = 0; c0 < nk; c0 += 32) {
    // scores: two 16x16 tiles (keys c0..+15, c0+16..+31), K = 64
    v8f s0 = {}, s1 = {};
    s0 = wmma32(aq0, load_b_nt(k, 64, c0, 0),       s0);
    s0 = wmma32(aq1, load_b_nt(k, 64, c0, 32),      s0);
    s1 = wmma32(aq0, load_b_nt(k, 64, c0 + 16, 0),  s1);
    s1 = wmma32(aq1, load_b_nt(k, 64, c0 + 16, 32), s1);

    float scl[8];
#pragma unroll
    for (int r = 0; r < 8; ++r) {
      float mx = fmaxf(s0[r], s1[r]);
      for (int off = 1; off < 16; off <<= 1)
        mx = fmaxf(mx, __shfl_xor(mx, off, 32));      // reduce within 16-lane half
      float mn = fmaxf(mrow[r], mx);
      float sc = __expf(mrow[r] - mn);
      float p0 = __expf(s0[r] - mn);
      float p1 = __expf(s1[r] - mn);
      float rs = p0 + p1;
      for (int off = 1; off < 16; off <<= 1)
        rs += __shfl_xor(rs, off, 32);
      lrow[r] = lrow[r] * sc + rs;
      mrow[r] = mn;
      scl[r] = sc;
      s0[r] = p0;
      s1[r] = p1;
    }
#pragma unroll
    for (int t = 0; t < 4; ++t)
#pragma unroll
      for (int r = 0; r < 8; ++r) acc[t][r] *= scl[r];

    // C-layout P -> LDS -> A-fragment layout
#pragma unroll
    for (int r = 0; r < 8; ++r) {
      lds_p[r + 8 * hi][ln]      = (h16)s0[r];
      lds_p[r + 8 * hi][16 + ln] = (h16)s1[r];
    }
    __syncthreads();
    v16h pa;
    {
      const h16* pr = &lds_p[ln][0];
      v8h plo = *(const v8h*)(pr + 16 * hi);
      v8h phi = *(const v8h*)(pr + 16 * hi + 8);
#pragma unroll
      for (int i = 0; i < 8; ++i) { pa[i] = plo[i]; pa[i + 8] = phi[i]; }
    }
    __syncthreads();

    // P(16x32) @ V(32x64): 4 output d-tiles, B from VT rows (contiguous keys)
#pragma unroll
    for (int t = 0; t < 4; ++t) {
      const h16* bp = vt + (size_t)(16 * t + ln) * nk + c0 + 16 * hi;
      v16h bv = *(const v16h*)bp;
      acc[t] = wmma32(pa, bv, acc[t]);
    }
  }

#pragma unroll
  for (int t = 0; t < 4; ++t)
#pragma unroll
    for (int r = 0; r < 8; ++r) {
      int m = m0 + r + 8 * hi, d = 16 * t + ln;
      float val = acc[t][r] / lrow[r];
      if (O)  O[(size_t)bh * nq * 64 + (size_t)m * 64 + d] = val;
      if (OT) OT[(size_t)bh * 64 * nq + (size_t)d * nq + m] = (h16)val;
    }
}

// attnC[b][i][h*64+d] = attn[bh][i][d] + depthwise_conv33(v)[bh][i][d]
__global__ void k_conv_merge(const float* __restrict__ attn,
                             const float* __restrict__ v32,
                             const float* __restrict__ cw,
                             h16* __restrict__ attnC) {
  int idx = blockIdx.x * 256 + threadIdx.x;       // 4*8*4096*64
  int d = idx & 63, i = (idx >> 6) & 4095, h = (idx >> 18) & 7, bb = idx >> 21;
  size_t base = ((size_t)(bb * 8 + h) * 4096) * 64 + d;
  float res = 0.f;
#pragma unroll
  for (int t = 0; t < 33; ++t) {
    int src = i + t - 16;
    if (src >= 0 && src < 4096) res += cw[h * 33 + t] * v32[base + (size_t)src * 64];
  }
  float val = attn[base + (size_t)i * 64] + res;
  attnC[((size_t)bb * 4096 + i) * 512 + h * 64 + d] = (h16)val;
}

// out = x*omega + attnC @ w_out + b_out (NT). 16x64 strip per wave.
// grid (1024, 8), 32 thr
__global__ void k_gemm_out(const h16* __restrict__ A, const h16* __restrict__ BT,
                           const float* __restrict__ x, const float* __restrict__ bout,
                           const float* __restrict__ omega, float* __restrict__ out) {
  int m0 = blockIdx.x * 16, n0 = blockIdx.y * 64;
  v8f acc[4] = {v8f{}, v8f{}, v8f{}, v8f{}};
#pragma unroll 2
  for (int k0 = 0; k0 < 512; k0 += 32) {
    __builtin_prefetch(A + (size_t)m0 * 512 + k0 + 128, 0, 0);
    v16h af = load_a_nt(A, 512, m0, k0);
#pragma unroll
    for (int t = 0; t < 4; ++t)
      acc[t] = wmma32(af, load_b_nt(BT, 512, n0 + 16 * t, k0), acc[t]);
  }
  int lane = threadIdx.x & 31, hi = lane >> 4;
  float om = omega[0];
#pragma unroll
  for (int t = 0; t < 4; ++t) {
    int n = n0 + 16 * t + (lane & 15);
    float bo = bout[n];
#pragma unroll
    for (int r = 0; r < 8; ++r) {
      size_t o = (size_t)(m0 + r + 8 * hi) * 512 + n;
      out[o] = x[o] * om + acc[t][r] + bo;
    }
  }
}

// ===========================================================================
extern "C" void kernel_launch(void* const* d_in, const int* in_sizes, int n_in,
                              void* d_out, int out_size, void* d_ws, size_t ws_size,
                              hipStream_t stream) {
  const float* x     = (const float*)d_in[0];
  const float* ln_w  = (const float*)d_in[1];
  const float* ln_b  = (const float*)d_in[2];
  const float* w_qkv = (const float*)d_in[3];
  const float* w_out = (const float*)d_in[4];
  const float* b_out = (const float*)d_in[5];
  const float* cw    = (const float*)d_in[6];
  const float* omega = (const float*)d_in[7];
  float* out = (float*)d_out;

  // ---- workspace layout (~210 MB, 256B aligned) ----
  char* p = (char*)d_ws;
  size_t off = 0;
  auto alloc = [&](size_t bytes) {
    void* r = p + off;
    off = (off + bytes + 255) & ~(size_t)255;
    return r;
  };
  h16*   nx     = (h16*)alloc((size_t)BN_ * 512 * 2);
  h16*   wqkvT  = (h16*)alloc((size_t)1536 * 512 * 2);
  h16*   woutT  = (h16*)alloc((size_t)512 * 512 * 2);
  h16*   qf     = (h16*)alloc((size_t)BH_ * N_ * 64 * 2);
  h16*   kf     = (h16*)alloc((size_t)BH_ * N_ * 64 * 2);
  h16*   vT     = (h16*)alloc((size_t)BH_ * 64 * N_ * 2);
  float* v32    = (float*)alloc((size_t)BH_ * N_ * 64 * 4);
  h16*   ql     = (h16*)alloc((size_t)BH_ * M_ * 64 * 2);
  h16*   kl     = (h16*)alloc((size_t)BH_ * M_ * 64 * 2);
  float* sim2   = (float*)alloc((size_t)BH_ * M_ * M_ * 4);
  h16*   a2     = (h16*)alloc((size_t)BH_ * M_ * M_ * 2);
  float* denom  = (float*)alloc(BH_ * 4);
  h16*   zA     = (h16*)alloc((size_t)BH_ * M_ * M_ * 2);
  h16*   zAT    = (h16*)alloc((size_t)BH_ * M_ * M_ * 2);
  h16*   zB     = (h16*)alloc((size_t)BH_ * M_ * M_ * 2);
  h16*   zBT    = (h16*)alloc((size_t)BH_ * M_ * M_ * 2);
  h16*   xz     = (h16*)alloc((size_t)BH_ * M_ * M_ * 2);
  h16*   yTa    = (h16*)alloc((size_t)BH_ * M_ * M_ * 2);
  h16*   yTb    = (h16*)alloc((size_t)BH_ * M_ * M_ * 2);
  h16*   a3vT   = (h16*)alloc((size_t)BH_ * 64 * M_ * 2);
  h16*   WT     = (h16*)alloc((size_t)BH_ * 64 * M_ * 2);
  float* attn   = (float*)alloc((size_t)BH_ * N_ * 64 * 4);
  h16*   attnC  = (h16*)alloc((size_t)BN_ * 512 * 2);
  (void)ws_size; (void)n_in; (void)in_sizes; (void)out_size;

  // 1. weight transposes + layernorm
  k_castT<<<(512 * 1536) / 256, 256, 0, stream>>>(w_qkv, wqkvT, 512, 1536);
  k_castT<<<(512 * 512) / 256, 256, 0, stream>>>(w_out, woutT, 512, 512);
  k_ln<<<BN_, 256, 0, stream>>>(x, ln_w, ln_b, nx);

  // 2. QKV projection (WMMA NT, 16x64 strip per wave)
  k_gemm_qkv<<<dim3(BN_ / 16, 1536 / 64), 32, 0, stream>>>(nx, wqkvT, qf, kf, v32, vT);

  // 3. landmarks
  k_landmarks<<<(BH_ * M_ * 64) / 256, 256, 0, stream>>>(qf, kf, ql, kl);

  // 4. a2 = softmax(ql @ kl^T); pinv init (z0 = a2^T / (max colsum * max rowsum))
  k_sim2<<<dim3(16, 16, BH_), 32, 0, stream>>>(ql, kl, sim2);
  k_softmax2<<<BH_ * M_, 256, 0, stream>>>(sim2, a2);
  k_denom<<<BH_, 256, 0, stream>>>(a2, denom);
  k_z0<<<(BH_ * M_ * M_) / 256, 256, 0, stream>>>(a2, denom, zA, zAT);

  // 5. Newton-Schulz: z = 0.25*z@(13I - xz@(15I - xz@(7I - xz))), xz = a2@z
  //    All NT: B operands consumed through transposed f16 copies written free
  //    in the producing epilogue.
  h16 *zc = zA, *zcT = zAT, *zn = zB, *znT = zBT;
  dim3 g256(16, 16, BH_);
  for (int it = 0; it < 6; ++it) {
    // xz = a2 @ z ; y1^T = (7I - xz)^T
    k_pinv_gemm<<<g256, 32, 0, stream>>>(a2, zcT, xz, (h16*)nullptr, 1.f, 0.f,
                                         yTa, -1.f, 7.f);
    // y3^T = (15I - xz @ y1)^T
    k_pinv_gemm<<<g256, 32, 0, stream>>>(xz, yTa, (h16*)nullptr, yTb, -1.f, 15.f,
                                         (h16*)nullptr, 0.f, 0.f);
    // y5^T = (13I - xz @ y3)^T
    k_pinv_gemm<<<g256, 32, 0, stream>>>(xz, yTb, (h16*)nullptr, yTa, -1.f, 13.f,
                                         (h16*)nullptr, 0.f, 0.f);
    // z' = 0.25 * z @ y5  (both row-major and transposed for next iteration)
    k_pinv_gemm<<<g256, 32, 0, stream>>>(zc, yTa, zn, znT, 0.25f, 0.f,
                                         (h16*)nullptr, 0.f, 0.f);
    h16* t;
    t = zc; zc = zn; zn = t;
    t = zcT; zcT = znT; znT = t;
  }

  // 6. a3v^T = (softmax(ql @ k^T) @ v)^T  (fused flash, nq=256, nk=4096)
  k_flash<<<dim3(M_ / 16, BH_), 32, 0, stream>>>(ql, kf, vT, (float*)nullptr,
                                                 a3vT, M_, N_);

  // 7. W^T = (pinv(a2) @ a3v)^T   (NT via a3v^T)
  k_w_gemm<<<dim3(16, 4, BH_), 32, 0, stream>>>(zc, a3vT, WT);

  // 8. nystrom out = softmax(q @ kl^T) @ W   (fused flash, nq=4096, nk=256)
  k_flash<<<dim3(N_ / 16, BH_), 32, 0, stream>>>(qf, kl, WT, attn, (h16*)nullptr,
                                                 N_, M_);

  // 9. + depthwise conv residual, merge heads -> [b][n][512] f16
  k_conv_merge<<<(BH_ * N_ * 64) / 256, 256, 0, stream>>>(attn, v32, cw, attnC);

  // 10. out-proj + bias + admin residual (x*omega + ...)
  k_gemm_out<<<dim3(BN_ / 16, 512 / 64), 32, 0, stream>>>(attnC, woutT, x, b_out,
                                                          omega, out);
}
